// STGCNSpatialConv_36971078484449
// MI455X (gfx1250) — compile-verified, matrix operand order checked by reading, not run on hardware
//
#include <hip/hip_runtime.h>
#include <hip/hip_bf16.h>
#include <math.h>

// ---------------- problem constants ----------------
constexpr int B_ = 8, T_ = 16, N_ = 4096, C_ = 64;
constexpr int NK_ = 96, K_ = 125, M_ = 32;
constexpr int E_ = NK_ + K_;    // 221
constexpr int EP_ = 224;        // padded to 14 * 16 for WMMA tiles
constexpr float GAMMA_ = 0.5f;
constexpr float EPS_ = 1e-12f;

// ---------------- WMMA fragment machinery (wave32, 16x16x32 f16) ------------
typedef __attribute__((ext_vector_type(16))) _Float16 v16h;
typedef __attribute__((ext_vector_type(8)))  float    v8f;

static __device__ inline v8f wmma_f16(v16h a, v16h b, v8f c) {
  // (neg_a, A, neg_b, B, c_mod, C, reuse_a, reuse_b)
  return __builtin_amdgcn_wmma_f32_16x16x32_f16(false, a, false, b, (short)0, c,
                                                false, false);
}

// Fragment storage: 32 lanes x 16 contiguous halves (1KB per 16x32 fragment).
// Forward layout (ISA 7.12.2): lane l -> mn = l&15, h = l>>4;
//   elem e<8  -> k = 8h+e ; elem e>=8 -> k = 16+8h+(e-8)
static __device__ inline void frag_put(_Float16* frag, int mn, int k, float v) {
  const int hi = k >> 4;
  const int w  = k & 15;
  const int lane = (mn & 15) + ((w >> 3) << 4);
  const int e    = (w & 7) + (hi << 3);
  frag[lane * 16 + e] = (_Float16)v;
}
// Contiguous 32B per-lane fragment load -> b128 pairs
static __device__ inline v16h frag_get(const _Float16* frag, int lane) {
  return *(const v16h*)(frag + lane * 16);
}

// ---------------- kernel 0: weights -> B-fragment f16 layout ---------------
__global__ void stgcn_wfrag(const float* __restrict__ lowW,
                            const float* __restrict__ upW,
                            _Float16* __restrict__ lowWf,
                            _Float16* __restrict__ upWf) {
  const int t = blockIdx.x * 256 + threadIdx.x;
  if (t >= C_ * C_) return;
  const int c = t / C_;                  // K index
  const int f = t % C_;                  // N index
  const int fi = (c >> 5) * 4 + (f >> 4);
  frag_put(lowWf + (size_t)fi * 512, f & 15, c & 31, lowW[t]);
  frag_put(upWf  + (size_t)fi * 512, f & 15, c & 31, upW[t]);
}

// ---------------- kernel 1: sym_norm(adj[cidx,cidx]) -> A-fragments --------
__global__ void stgcn_asubn(const float* __restrict__ adj,
                            const int* __restrict__ cidx,
                            _Float16* __restrict__ asubnf) {
  const int k = blockIdx.x;              // 0..K-1
  const int l = threadIdx.x;             // 0..31, one row per lane
  __shared__ float a[M_][M_];
  __shared__ float dinv[M_];
  const int* gk = cidx + k * M_;
  const int row = gk[l];
  float d = 0.f;
  for (int j = 0; j < M_; ++j) {
    float v = adj[(size_t)row * N_ + gk[j]] + (j == l ? 1.f : 0.f);
    a[l][j] = v;
    d += v;
  }
  dinv[l] = d > 0.f ? rsqrtf(d) : 0.f;
  __syncthreads();
  _Float16* base = asubnf + (size_t)k * 2 * 512 + (size_t)(l >> 4) * 512;
  for (int j = 0; j < M_; ++j)
    frag_put(base, l & 15, j, a[l][j] * dinv[l] * dinv[j]);
}

// ---------------- kernel 2: w = memb @ adj @ memb^T ------------------------
__global__ void stgcn_condw(const float* __restrict__ adj,
                            const int* __restrict__ kidx,
                            const int* __restrict__ cidx,
                            float* __restrict__ wbuf) {
  const int t = blockIdx.x * 256 + threadIdx.x;
  if (t >= E_ * E_) return;
  const int i = t / E_, j = t % E_;
  float s = 0.f;
  if (i < NK_) {
    const size_t rb = (size_t)kidx[i] * N_;
    if (j < NK_) {
      s = adj[rb + kidx[j]];
    } else {
      const int* gj = cidx + (j - NK_) * M_;
      for (int b = 0; b < M_; ++b) s += adj[rb + gj[b]];
    }
  } else {
    const int* gi = cidx + (i - NK_) * M_;
    if (j < NK_) {
      const int q = kidx[j];
      for (int a = 0; a < M_; ++a) s += adj[(size_t)gi[a] * N_ + q];
    } else {
      const int* gj = cidx + (j - NK_) * M_;
      for (int a = 0; a < M_; ++a) {
        const size_t rb = (size_t)gi[a] * N_;
        for (int b = 0; b < M_; ++b) s += adj[rb + gj[b]];
      }
    }
  }
  wbuf[t] = s;
}

// -------- kernel 3: An = sym_norm((w>0)*(1-I)) -> padded A-fragments -------
__global__ void stgcn_condnorm(const float* __restrict__ wbuf,
                               _Float16* __restrict__ Anf) {
  __shared__ float dinv[EP_];
  const int tid = threadIdx.x;           // 256 threads, 1 block
  for (int i = tid; i < EP_; i += 256) {
    float d = 0.f;
    if (i < E_) {
      for (int j = 0; j < E_; ++j)
        d += (i == j) ? 1.f : ((wbuf[i * E_ + j] > 0.f) ? 1.f : 0.f);
    }
    dinv[i] = d > 0.f ? rsqrtf(d) : 0.f;
  }
  __syncthreads();
  for (int t = tid; t < EP_ * EP_; t += 256) {
    const int i = t / EP_, j = t % EP_;
    float v = 0.f;
    if (i < E_ && j < E_) {
      float aij = (i == j) ? 1.f : ((wbuf[i * E_ + j] > 0.f) ? 1.f : 0.f);
      v = aij * dinv[i] * dinv[j];
    }
    _Float16* f = Anf + ((size_t)(i >> 4) * (EP_ / 32) + (j >> 5)) * 512;
    frag_put(f, i & 15, j & 31, v);
  }
}

// ---------------- kernel 4: per (b,t,k) cluster refine + attention pool ----
// One wave per block. WMMA phases are batched: all independent accumulator
// tiles first (no D->A/B hazards between them, so they pipeline), then the
// conversion/scatter epilogue, avoiding per-tile WMMA->VALU hazard NOPs.
__global__ void stgcn_cluster(const float* __restrict__ x,
                              const int* __restrict__ cidx,
                              const float* __restrict__ prev,
                              const _Float16* __restrict__ asubnf,
                              const _Float16* __restrict__ lowWf,
                              const float* __restrict__ lowb,
                              const float* __restrict__ qW,
                              float* __restrict__ out_full,
                              float* __restrict__ out_pooled) {
  const int bid = blockIdx.x;
  const int k = bid % K_;
  const int bt = bid / K_;
  const int lane = threadIdx.x;          // 0..31

  __shared__ __align__(32) _Float16 sXf[4][512];     // xs  B-fragments (4KB)
  __shared__ __align__(32) _Float16 sGf[2][2][512];  // agg A-fragments (4KB)
  __shared__ float sR[M_][C_];                       // refined f32 (8KB)
  __shared__ float sq[C_];
  __shared__ float slog[M_];
  __shared__ float sattn[M_];

  const int* gk = cidx + k * M_;
  const float* xbt = x + (size_t)bt * N_ * C_;

  // gather xs row `lane` with b128 loads, scatter into B-fragment layout
  {
    const float4* xr = (const float4*)(xbt + (size_t)gk[lane] * C_);
#pragma unroll
    for (int cc = 0; cc < C_ / 4; ++cc) {
      const float4 v = xr[cc];
      const int c0 = cc * 4;
      frag_put(&sXf[(c0 + 0) >> 4][0], (c0 + 0) & 15, lane, v.x);
      frag_put(&sXf[(c0 + 1) >> 4][0], (c0 + 1) & 15, lane, v.y);
      frag_put(&sXf[(c0 + 2) >> 4][0], (c0 + 2) & 15, lane, v.z);
      frag_put(&sXf[(c0 + 3) >> 4][0], (c0 + 3) & 15, lane, v.w);
    }
  }
  // A_sub_n fragments + lower_W fragments cached in VGPRs (global b128 loads)
  v16h aA0 = frag_get(asubnf + (size_t)k * 2 * 512, lane);
  v16h aA1 = frag_get(asubnf + (size_t)k * 2 * 512 + 512, lane);
  v16h wf[2][4];
#pragma unroll
  for (int kk = 0; kk < 2; ++kk)
#pragma unroll
    for (int tj = 0; tj < 4; ++tj)
      wf[kk][tj] = frag_get(lowWf + (size_t)(kk * 4 + tj) * 512, lane);
  __syncthreads();

  const int hC = (lane >> 4) & 1, nC = lane & 15;

  // ---- agg = A_sub_n(32x32) @ xs(32x64): batch all 8 WMMAs, then scatter ----
  v8f ga[2][4];
#pragma unroll
  for (int ti = 0; ti < 2; ++ti)
#pragma unroll
    for (int tj = 0; tj < 4; ++tj) {
      v8f z = {};
      ga[ti][tj] = wmma_f16(ti == 0 ? aA0 : aA1, frag_get(&sXf[tj][0], lane), z);
    }
#pragma unroll
  for (int ti = 0; ti < 2; ++ti)
#pragma unroll
    for (int tj = 0; tj < 4; ++tj)
#pragma unroll
      for (int r = 0; r < 8; ++r) {
        const int row = r + 8 * hC;
        const int col = tj * 16 + nC;
        frag_put(&sGf[ti][col >> 5][0], row, col & 31, ga[ti][tj][r]);
      }
  __syncthreads();

  // ---- refined = relu(agg @ lower_W + b): batch all 8 tiles, then epilogue --
  v8f rf[2][4];
#pragma unroll
  for (int ti = 0; ti < 2; ++ti) {
    const v16h g0 = frag_get(&sGf[ti][0][0], lane);
    const v16h g1 = frag_get(&sGf[ti][1][0], lane);
#pragma unroll
    for (int tj = 0; tj < 4; ++tj) {
      v8f acc = {};
      acc = wmma_f16(g0, wf[0][tj], acc);
      rf[ti][tj] = wmma_f16(g1, wf[1][tj], acc);
    }
  }
#pragma unroll
  for (int ti = 0; ti < 2; ++ti)
#pragma unroll
    for (int tj = 0; tj < 4; ++tj) {
      const float bias = lowb[tj * 16 + nC];
#pragma unroll
      for (int r = 0; r < 8; ++r) {
        float v = rf[ti][tj][r] + bias;
        v = v > 0.f ? v : 0.f;
        sR[ti * 16 + r + 8 * hC][tj * 16 + nC] = v;
      }
    }
  __syncthreads();

  // q = l2norm(prev[bt,k,:] @ query_W)   (each lane: 2 output features)
  const float* p = prev + ((size_t)bt * K_ + k) * C_;
  float q0 = 0.f, q1 = 0.f;
  for (int c = 0; c < C_; ++c) {
    const float pv = p[c];
    q0 = fmaf(pv, qW[c * C_ + lane], q0);
    q1 = fmaf(pv, qW[c * C_ + lane + 32], q1);
  }
  sq[lane] = q0; sq[lane + 32] = q1;
  __syncthreads();
  float qn = 0.f;
  for (int c = 0; c < C_; ++c) qn += sq[c] * sq[c];
  const float qinv = 1.f / fmaxf(sqrtf(qn), EPS_);
  __syncthreads();
  sq[lane] = q0 * qinv; sq[lane + 32] = q1 * qinv;
  __syncthreads();

  // logits[m] = <l2norm(refined[m]), q>   (lane = row m)
  {
    float nrm = 0.f, dot = 0.f;
    for (int c = 0; c < C_; ++c) {
      const float v = sR[lane][c];
      nrm += v * v;
      dot = fmaf(v, sq[c], dot);
    }
    slog[lane] = dot / fmaxf(sqrtf(nrm), EPS_);
  }
  __syncthreads();
  {
    float mx = -INFINITY;
    for (int m = 0; m < M_; ++m) mx = fmaxf(mx, slog[m]);
    sattn[lane] = expf(slog[lane] - mx);
  }
  __syncthreads();
  {
    float ssum = 0.f;
    for (int m = 0; m < M_; ++m) ssum += sattn[m];
    const float inv = 1.f / ssum;
    float p0 = 0.f, p1 = 0.f;
    for (int m = 0; m < M_; ++m) {
      const float w = sattn[m] * inv;
      p0 = fmaf(w, sR[m][lane], p0);
      p1 = fmaf(w, sR[m][lane + 32], p1);
    }
    float* pout = out_pooled + ((size_t)bt * K_ + k) * C_;
    pout[lane] = p0;
    pout[lane + 32] = p1;
  }

  // scatter (1-GAMMA)*refined straight into the full output (lane = row)
  {
    float4* frow = (float4*)(out_full + ((size_t)bt * N_ + gk[lane]) * C_);
#pragma unroll
    for (int cc = 0; cc < C_ / 4; ++cc) {
      float4 v;
      v.x = (1.f - GAMMA_) * sR[lane][cc * 4 + 0];
      v.y = (1.f - GAMMA_) * sR[lane][cc * 4 + 1];
      v.z = (1.f - GAMMA_) * sR[lane][cc * 4 + 2];
      v.w = (1.f - GAMMA_) * sR[lane][cc * 4 + 3];
      frow[cc] = v;
    }
  }
}

// ---------------- kernel 5: cond stage per (b,t) ---------------------------
// 8 waves; wave w handles row tiles w, w+8. A-fragments of An from global.
// First-matmul WMMAs batched across tj before the fragment scatter.
__global__ void stgcn_cond(const float* __restrict__ x,
                           const float* __restrict__ pooled,
                           const int* __restrict__ kidx,
                           const _Float16* __restrict__ Anf,
                           const _Float16* __restrict__ upWf,
                           const float* __restrict__ upb,
                           float* __restrict__ condout) {
  const int bt = blockIdx.x;
  const int tid = threadIdx.x;           // 0..255
  const int lane = tid & 31;
  const int wave = tid >> 5;             // 0..7

  __shared__ __align__(32) _Float16 sCXf[7][4][512];  // cond_x B-frags (28KB)
  __shared__ __align__(32) _Float16 sTmpf[8][2][512]; // per-wave tmp A-frags (16KB)
  __shared__ float sb2[C_];

  const float* xbt = x + (size_t)bt * N_ * C_;
  const float* pbt = pooled + (size_t)bt * K_ * C_;

  for (int t = tid; t < EP_ * C_; t += 256) {
    const int i = t / C_, c = t % C_;
    float v = 0.f;
    if (i < NK_) v = xbt[(size_t)kidx[i] * C_ + c];
    else if (i < E_) v = pbt[(size_t)(i - NK_) * C_ + c];
    frag_put(&sCXf[i >> 5][c >> 4][0], c & 15, i & 31, v);
  }
  for (int t = tid; t < C_; t += 256) sb2[t] = upb[t];

  // cache upper_W fragments in VGPRs
  v16h w2[2][4];
#pragma unroll
  for (int kk = 0; kk < 2; ++kk)
#pragma unroll
    for (int tj = 0; tj < 4; ++tj)
      w2[kk][tj] = frag_get(upWf + (size_t)(kk * 4 + tj) * 512, lane);
  __syncthreads();

  const int hC = (lane >> 4) & 1, nC = lane & 15;
  for (int ti = wave; ti < EP_ / 14 * 14 / 16; /*14 tiles*/ ti += 8) {
    // hoist the 7 A-fragments of this row strip (global b128 loads, L2-hot)
    v16h af[7];
#pragma unroll
    for (int kk = 0; kk < 7; ++kk)
      af[kk] = frag_get(Anf + ((size_t)ti * 7 + kk) * 512, lane);
    // tmp = An @ cond_x for this strip: batch all 4 tiles, then scatter
    v8f ta[4];
#pragma unroll
    for (int tj = 0; tj < 4; ++tj) {
      v8f acc = {};
#pragma unroll
      for (int kk = 0; kk < 7; ++kk)
        acc = wmma_f16(af[kk], frag_get(&sCXf[kk][tj][0], lane), acc);
      ta[tj] = acc;
    }
#pragma unroll
    for (int tj = 0; tj < 4; ++tj)
#pragma unroll
      for (int r = 0; r < 8; ++r) {
        const int row = r + 8 * hC;
        const int col = tj * 16 + nC;
        frag_put(&sTmpf[wave][col >> 5][0], row, col & 31, ta[tj][r]);
      }
    // cond_out = relu(tmp @ upper_W + b): batch 4 tiles, then epilogue
    const v16h t0 = frag_get(&sTmpf[wave][0][0], lane);
    const v16h t1 = frag_get(&sTmpf[wave][1][0], lane);
    v8f oa[4];
#pragma unroll
    for (int tj = 0; tj < 4; ++tj) {
      v8f acc = {};
      acc = wmma_f16(t0, w2[0][tj], acc);
      oa[tj] = wmma_f16(t1, w2[1][tj], acc);
    }
#pragma unroll
    for (int tj = 0; tj < 4; ++tj) {
      const int col = tj * 16 + nC;
      const float bias = sb2[col];
#pragma unroll
      for (int r = 0; r < 8; ++r) {
        const int row = ti * 16 + r + 8 * hC;
        float v = oa[tj][r] + bias;
        v = v > 0.f ? v : 0.f;
        if (row < E_) condout[((size_t)bt * E_ + row) * C_ + col] = v;
      }
    }
  }
}

// ---------------- kernel 6: final scatter / fuse ---------------------------
__global__ void stgcn_fuse(const int* __restrict__ kidx,
                           const int* __restrict__ cidx,
                           const float* __restrict__ condout,
                           float* __restrict__ out_full) {
  const size_t t = (size_t)blockIdx.x * 256 + threadIdx.x;
  const size_t total = (size_t)B_ * T_ * N_ * C_;
  if (t >= total) return;
  const int c = (int)(t % C_);
  const size_t s = t / C_;
  const int slot = (int)(s % N_);
  const int bt = (int)(s / N_);
  if (slot < NK_) {
    const int node = kidx[slot];
    out_full[((size_t)bt * N_ + node) * C_ + c] =
        condout[((size_t)bt * E_ + slot) * C_ + c];
  } else {
    const int sp = slot - NK_;
    const int k = sp / M_;
    const int m = sp % M_;
    const int node = cidx[k * M_ + m];
    const size_t o = ((size_t)bt * N_ + node) * C_ + c;
    out_full[o] += GAMMA_ * condout[((size_t)bt * E_ + NK_ + k) * C_ + c];
  }
}

// ---------------- host launcher ----------------
extern "C" void kernel_launch(void* const* d_in, const int* in_sizes, int n_in,
                              void* d_out, int out_size, void* d_ws, size_t ws_size,
                              hipStream_t stream) {
  (void)in_sizes; (void)n_in; (void)out_size; (void)ws_size;

  const float* x    = (const float*)d_in[0];
  const float* adj  = (const float*)d_in[1];
  const int*   kidx = (const int*)d_in[2];
  const int*   cidx = (const int*)d_in[3];
  const float* prev = (const float*)d_in[4];
  const float* lowW = (const float*)d_in[5];
  const float* lowb = (const float*)d_in[6];
  const float* upW  = (const float*)d_in[7];
  const float* upb  = (const float*)d_in[8];
  const float* qW   = (const float*)d_in[9];

  float* out_full   = (float*)d_out;
  float* out_pooled = out_full + (size_t)B_ * T_ * N_ * C_;

  // workspace layout (all regions 32B aligned)
  _Float16* asubnf = (_Float16*)d_ws;               // K*2*512     = 128000 halves
  _Float16* Anf    = asubnf + (size_t)K_ * 2 * 512; // 14*7*512    =  50176 halves
  _Float16* lowWf  = Anf + (size_t)14 * 7 * 512;    // 4096 halves
  _Float16* upWf   = lowWf + 4096;                  // 4096 halves
  float*    wbuf   = (float*)(upWf + 4096);         // E*E (padded to 50176)
  float*    condout = wbuf + 50176;                 // B*T*E*C = 1810432 floats

  stgcn_wfrag<<<(C_ * C_ + 255) / 256, 256, 0, stream>>>(lowW, upW, lowWf, upWf);
  stgcn_asubn<<<K_, 32, 0, stream>>>(adj, cidx, asubnf);
  stgcn_condw<<<(E_ * E_ + 255) / 256, 256, 0, stream>>>(adj, kidx, cidx, wbuf);
  stgcn_condnorm<<<1, 256, 0, stream>>>(wbuf, Anf);
  stgcn_cluster<<<B_ * T_ * K_, 32, 0, stream>>>(x, cidx, prev, asubnf, lowWf,
                                                 lowb, qW, out_full, out_pooled);
  stgcn_cond<<<B_ * T_, 256, 0, stream>>>(x, out_pooled, kidx, Anf, upWf, upb,
                                          condout);
  const size_t total = (size_t)B_ * T_ * N_ * C_;
  stgcn_fuse<<<(unsigned)((total + 255) / 256), 256, 0, stream>>>(kidx, cidx,
                                                                  condout, out_full);
}
